// ModifiedDecoder_7602092113961
// MI455X (gfx1250) — compile-verified, matrix operand order checked by reading, not run on hardware
//
#include <hip/hip_runtime.h>
#include <hip/hip_bf16.h>

// ---------------- model constants ----------------
#define LNUM 6
#define DMODEL 768
#define NHEAD 12
#define HDIM 64
#define VOCAB 50257
#define SMAX 1024
#define RLORA 16
#define BATCH 2
#define TIMG 257
#define TTXT 767
#define TSEQ 1024              // 257 + 767, == SMAX
#define LSCALE (1.0f / 16.0f)  // LoRA alpha/r

typedef __attribute__((ext_vector_type(16))) _Float16 v16h;
typedef __attribute__((ext_vector_type(8)))  float    v8f;

union FragH { v16h v; unsigned u[8]; };
union FragF { v8f  v; float    f[8]; };
union Pack4 { _Float16 h[4]; unsigned long long u; };

// ---------------- WMMA GEMM ----------------------
// C[m][n] = alpha * sum_k A[m][k]*B(k,n)  (+bias[n]) (+resid[m][n])
//           (+ lscale * sum_j loraL[m][j]*loraR[n][j])
// A row-major MxK (lda). B: TRANSB ? row-major NxK (ldb) : row-major KxN (ldb).
// Assumptions (hold for every call site below): K % 32 == 0; for !TRANSB, N is a
// multiple of 64 (no ragged columns); ragged edges only on M and on TRANSB-N,
// both handled with clamped loads + select-zero (branch-free).
// Batch offset for X in {A,B,C}: (z/batchH)*xOuter + (z%batchH)*xInner.
#define TM 128
#define TN 64
#define TK 32
#define LDSH 40   // 32 + 8 halfword pad

template <int TRANSB>
__global__ __launch_bounds__(256)
void gemm_wmma(const float* __restrict__ A, const float* __restrict__ B,
               const float* __restrict__ bias, const float* __restrict__ resid,
               const float* __restrict__ loraL, const float* __restrict__ loraR,
               float* __restrict__ C,
               int M, int N, int K, int lda, int ldb, int ldc,
               float alpha, float lscale, int batchH,
               long aOuter, long aInner, long bOuter, long bInner,
               long cOuter, long cInner)
{
    const int z = blockIdx.z;
    const long aOff = (long)(z / batchH) * aOuter + (long)(z % batchH) * aInner;
    const long bOff = (long)(z / batchH) * bOuter + (long)(z % batchH) * bInner;
    const long cOff = (long)(z / batchH) * cOuter + (long)(z % batchH) * cInner;
    A += aOff; B += bOff; C += cOff;
    if (resid) resid += cOff;

    const int m0 = blockIdx.y * TM;
    const int n0 = blockIdx.x * TN;
    const int tid  = threadIdx.x;
    const int lane = tid & 31;
    const int w    = tid >> 5;
    const int wm = (w & 3) << 5;   // wave row offset in tile (0,32,64,96)
    const int wn = (w >> 2) << 5;  // wave col offset (0,32)

    __shared__ __align__(16) _Float16 Ash[TM * LDSH];
    __shared__ __align__(16) _Float16 Bsh[TN * LDSH];   // stored transposed: [n][k]

    FragF acc00, acc01, acc10, acc11;
    #pragma unroll
    for (int i = 0; i < 8; ++i) {
        acc00.f[i] = 0.f; acc01.f[i] = 0.f; acc10.f[i] = 0.f; acc11.f[i] = 0.f;
    }

    const int hl  = lane & 15;
    const int kb2 = (lane >> 4) << 2;  // (lane<16 ? 0 : 8) halfwords -> dwords

    // per-thread staging coordinates (fixed across K-steps)
    // A: 4 float4 per thread (TM*TK/4 = 1024 over 256 threads)
    int arr[4], acc4[4];
    #pragma unroll
    for (int t = 0; t < 4; ++t) { int j = tid + t * 256; arr[t] = j >> 3; acc4[t] = j & 7; }
    // B: 2 float4 per thread (TN*TK/4 = 512 over 256 threads)
    int br[2], bc[2];
    #pragma unroll
    for (int t = 0; t < 2; ++t) {
        int j = tid + t * 256;
        if (TRANSB) { br[t] = j >> 3;  bc[t] = j & 7;  }   // n = j/8,  k4 = j%8
        else        { br[t] = j >> 4;  bc[t] = j & 15; }   // k = j/16, n4 = j%16
    }

    float4 regA[4], regB[2];

    auto loadA = [&](int k0) {
        #pragma unroll
        for (int t = 0; t < 4; ++t) {
            int gm = m0 + arr[t];
            int gmc = gm < M ? gm : M - 1;
            float4 v = *(const float4*)(A + (long)gmc * lda + k0 + acc4[t] * 4);
            if (gm >= M) v = make_float4(0.f, 0.f, 0.f, 0.f);
            regA[t] = v;
        }
    };
    auto loadB = [&](int k0) {
        #pragma unroll
        for (int t = 0; t < 2; ++t) {
            if (TRANSB) {
                int gn = n0 + br[t];
                int gnc = gn < N ? gn : N - 1;
                float4 v = *(const float4*)(B + (long)gnc * ldb + k0 + bc[t] * 4);
                if (gn >= N) v = make_float4(0.f, 0.f, 0.f, 0.f);
                regB[t] = v;
            } else {
                int gk = k0 + br[t];
                regB[t] = *(const float4*)(B + (long)gk * ldb + n0 + bc[t] * 4);
            }
        }
    };
    auto storeA = [&]() {
        #pragma unroll
        for (int t = 0; t < 4; ++t) {
            Pack4 p;
            p.h[0] = (_Float16)regA[t].x; p.h[1] = (_Float16)regA[t].y;
            p.h[2] = (_Float16)regA[t].z; p.h[3] = (_Float16)regA[t].w;
            *(unsigned long long*)(Ash + arr[t] * LDSH + acc4[t] * 4) = p.u;
        }
    };
    auto storeB = [&]() {
        #pragma unroll
        for (int t = 0; t < 2; ++t) {
            if (TRANSB) {
                Pack4 p;
                p.h[0] = (_Float16)regB[t].x; p.h[1] = (_Float16)regB[t].y;
                p.h[2] = (_Float16)regB[t].z; p.h[3] = (_Float16)regB[t].w;
                *(unsigned long long*)(Bsh + br[t] * LDSH + bc[t] * 4) = p.u;
            } else {
                int n4 = bc[t] * 4, k = br[t];
                Bsh[(n4 + 0) * LDSH + k] = (_Float16)regB[t].x;
                Bsh[(n4 + 1) * LDSH + k] = (_Float16)regB[t].y;
                Bsh[(n4 + 2) * LDSH + k] = (_Float16)regB[t].z;
                Bsh[(n4 + 3) * LDSH + k] = (_Float16)regB[t].w;
            }
        }
    };

    loadA(0); loadB(0);
    for (int k0 = 0; k0 < K; k0 += TK) {
        storeA(); storeB();
        __syncthreads();
        if (k0 + TK < K) { loadA(k0 + TK); loadB(k0 + TK); }  // overlap next tile

        FragH fa0, fa1, fb0, fb1;
        {
            const unsigned* a0 = (const unsigned*)(Ash + (wm + hl) * LDSH);
            const unsigned* a1 = (const unsigned*)(Ash + (wm + 16 + hl) * LDSH);
            const unsigned* b0 = (const unsigned*)(Bsh + (wn + hl) * LDSH);
            const unsigned* b1 = (const unsigned*)(Bsh + (wn + 16 + hl) * LDSH);
            #pragma unroll
            for (int i = 0; i < 4; ++i) {
                fa0.u[i] = a0[kb2 + i]; fa0.u[4 + i] = a0[8 + kb2 + i];
                fa1.u[i] = a1[kb2 + i]; fa1.u[4 + i] = a1[8 + kb2 + i];
                fb0.u[i] = b0[kb2 + i]; fb0.u[4 + i] = b0[8 + kb2 + i];
                fb1.u[i] = b1[kb2 + i]; fb1.u[4 + i] = b1[8 + kb2 + i];
            }
        }
        acc00.v = __builtin_amdgcn_wmma_f32_16x16x32_f16(false, fa0.v, false, fb0.v,
                                                         (short)0, acc00.v, false, false);
        acc01.v = __builtin_amdgcn_wmma_f32_16x16x32_f16(false, fa0.v, false, fb1.v,
                                                         (short)0, acc01.v, false, false);
        acc10.v = __builtin_amdgcn_wmma_f32_16x16x32_f16(false, fa1.v, false, fb0.v,
                                                         (short)0, acc10.v, false, false);
        acc11.v = __builtin_amdgcn_wmma_f32_16x16x32_f16(false, fa1.v, false, fb1.v,
                                                         (short)0, acc11.v, false, false);
        __syncthreads();
    }

    // epilogue: C/D layout — lane<16: N=lane, M=r; lane>=16: N=lane-16, M=8+r
    const int mb = (lane >> 4) << 3;
    #pragma unroll
    for (int mi = 0; mi < 2; ++mi) {
        #pragma unroll
        for (int ni = 0; ni < 2; ++ni) {
            const FragF& acc = mi ? (ni ? acc11 : acc10) : (ni ? acc01 : acc00);
            int gn = n0 + wn + ni * 16 + hl;
            if (gn >= N) continue;
            #pragma unroll
            for (int r = 0; r < 8; ++r) {
                int gm = m0 + wm + mi * 16 + mb + r;
                if (gm >= M) continue;
                float vv = acc.f[r] * alpha;
                if (bias)  vv += bias[gn];
                if (resid) vv += resid[(long)gm * ldc + gn];
                if (loraL) {
                    float s = 0.f;
                    #pragma unroll
                    for (int j = 0; j < RLORA; ++j)
                        s += loraL[(long)gm * RLORA + j] * loraR[(long)gn * RLORA + j];
                    vv += s * lscale;
                }
                C[(long)gm * ldc + gn] = vv;
            }
        }
    }
}

// ---------------- LayerNorm ----------------------
__global__ __launch_bounds__(256)
void ln_kernel(const float* __restrict__ x, const float* __restrict__ g,
               const float* __restrict__ b, float* __restrict__ y, int D)
{
    const int row = blockIdx.x;
    const float* xr = x + (long)row * D;
    float* yr = y + (long)row * D;
    __shared__ float r1[256], r2[256];
    int tid = threadIdx.x;
    float s = 0.f, sq = 0.f;
    for (int i = tid; i < D; i += 256) { float v = xr[i]; s += v; sq += v * v; }
    r1[tid] = s; r2[tid] = sq; __syncthreads();
    for (int off = 128; off > 0; off >>= 1) {
        if (tid < off) { r1[tid] += r1[tid + off]; r2[tid] += r2[tid + off]; }
        __syncthreads();
    }
    float mean = r1[0] / D;
    float var  = r2[0] / D - mean * mean;
    float inv  = rsqrtf(var + 1e-5f);
    for (int i = tid; i < D; i += 256)
        yr[i] = (xr[i] - mean) * inv * g[i] + b[i];
}

// ---------------- causal softmax -----------------
__global__ __launch_bounds__(256)
void softmax_causal(const float* __restrict__ scores, float* __restrict__ probs, int T)
{
    const int q  = blockIdx.x;
    const long base = ((long)blockIdx.y * T + q) * (long)T;
    const float* s = scores + base;
    float* p = probs + base;
    const int n = q + 1;  // causal: keys 0..q
    __shared__ float red[256];
    int tid = threadIdx.x;
    float m = -3.4e38f;
    for (int k = tid; k < n; k += 256) m = fmaxf(m, s[k]);
    red[tid] = m; __syncthreads();
    for (int off = 128; off > 0; off >>= 1) {
        if (tid < off) red[tid] = fmaxf(red[tid], red[tid + off]);
        __syncthreads();
    }
    m = red[0]; __syncthreads();
    float sum = 0.f;
    for (int k = tid; k < n; k += 256) sum += __expf(s[k] - m);
    red[tid] = sum; __syncthreads();
    for (int off = 128; off > 0; off >>= 1) {
        if (tid < off) red[tid] += red[tid + off];
        __syncthreads();
    }
    float inv = 1.f / red[0];
    for (int k = tid; k < n; k += 256) p[k] = __expf(s[k] - m) * inv;
    for (int k = n + tid; k < T; k += 256) p[k] = 0.f;
}

// ---------------- embedding assembly -------------
__global__ __launch_bounds__(256)
void assemble_h(const float* __restrict__ visln, const int* __restrict__ x,
                const float* __restrict__ wte, const float* __restrict__ wpe,
                float* __restrict__ h)
{
    long i = (long)blockIdx.x * 256 + threadIdx.x;
    const long total = (long)BATCH * TSEQ * DMODEL;
    if (i >= total) return;
    int d = (int)(i % DMODEL);
    long bt = i / DMODEL;
    int t = (int)(bt % TSEQ);
    int b = (int)(bt / TSEQ);
    float v;
    if (t < TIMG) v = visln[((long)b * TIMG + t) * DMODEL + d];
    else          v = wte[(long)x[(long)b * TTXT + (t - TIMG)] * DMODEL + d];
    h[i] = v + wpe[(long)t * DMODEL + d];
}

// ---------------- GELU (tanh approx) -------------
__global__ __launch_bounds__(256)
void gelu_kernel(float* __restrict__ a, long n)
{
    long i = (long)blockIdx.x * 256 + threadIdx.x;
    if (i >= n) return;
    float x = a[i];
    float u = 0.7978845608028654f * (x + 0.044715f * x * x * x);
    a[i] = 0.5f * x * (1.f + tanhf(u));
}

// ------------- LoRA down: out[M,16] = X[M,Dk] @ Aw[16,Dk]^T -------------
__global__ __launch_bounds__(256)
void lora_down(const float* __restrict__ X, const float* __restrict__ Aw,
               float* __restrict__ out, int M, int Dk)
{
    int i = blockIdx.x * 256 + threadIdx.x;
    if (i >= M * RLORA) return;
    int r = i & (RLORA - 1);
    int m = i >> 4;
    const float* xr = X + (long)m * Dk;
    const float* ar = Aw + (long)r * Dk;
    float s = 0.f;
    for (int k = 0; k < Dk; ++k) s += xr[k] * ar[k];
    out[i] = s;
}

// ---------------- host orchestration -------------
static inline void run_gemm(hipStream_t st, const float* A, const float* B,
                            const float* bias, const float* resid,
                            const float* lL, const float* lR, float* C,
                            int M, int N, int K, int lda, int ldb, int ldc,
                            int transB, float alpha, int batch, int batchH,
                            long aO, long aI, long bO, long bI, long cO, long cI)
{
    dim3 grid((N + TN - 1) / TN, (M + TM - 1) / TM, batch);
    if (transB)
        gemm_wmma<1><<<grid, 256, 0, st>>>(A, B, bias, resid, lL, lR, C,
                                           M, N, K, lda, ldb, ldc, alpha,
                                           LSCALE, batchH, aO, aI, bO, bI, cO, cI);
    else
        gemm_wmma<0><<<grid, 256, 0, st>>>(A, B, bias, resid, lL, lR, C,
                                           M, N, K, lda, ldb, ldc, alpha,
                                           LSCALE, batchH, aO, aI, bO, bI, cO, cI);
}

extern "C" void kernel_launch(void* const* d_in, const int* in_sizes, int n_in,
                              void* d_out, int out_size, void* d_ws, size_t ws_size,
                              hipStream_t stream)
{
    const int*   x      = (const int*)  d_in[0];
    const float* imgf   = (const float*)d_in[1];
    const float* wte    = (const float*)d_in[2];
    const float* wpe    = (const float*)d_in[3];
    const float* head_A = (const float*)d_in[4];
    const float* head_B = (const float*)d_in[5];
    const float* vp1_w  = (const float*)d_in[6];
    const float* vp1_b  = (const float*)d_in[7];
    const float* vp2_w  = (const float*)d_in[8];
    const float* vp2_b  = (const float*)d_in[9];
    const float* ln_g   = (const float*)d_in[10];
    const float* ln_b   = (const float*)d_in[11];
    const float* lnf_g  = (const float*)d_in[12];
    const float* lnf_b  = (const float*)d_in[13];
    const float* ln1_g  = (const float*)d_in[14];
    const float* ln1_b  = (const float*)d_in[15];
    const float* ln2_g  = (const float*)d_in[16];
    const float* ln2_b  = (const float*)d_in[17];
    const float* attn_w = (const float*)d_in[18];
    const float* attn_b = (const float*)d_in[19];
    const float* ap_w   = (const float*)d_in[20];
    const float* ap_b   = (const float*)d_in[21];
    const float* fc_w   = (const float*)d_in[22];
    const float* fc_b   = (const float*)d_in[23];
    const float* fc_A   = (const float*)d_in[24];
    const float* fc_B   = (const float*)d_in[25];
    const float* pj_w   = (const float*)d_in[26];
    const float* pj_b   = (const float*)d_in[27];
    const float* pj_A   = (const float*)d_in[28];
    const float* pj_B   = (const float*)d_in[29];

    float* out    = (float*)d_out;
    float* logits = out;                                   // [B,T,V]
    float* atts   = out + (long)BATCH * TSEQ * VOCAB;      // [L,B,H,T,T]

    const int M   = BATCH * TSEQ;   // 2048
    const int D   = DMODEL;
    const int D3  = 3 * DMODEL;
    const int D4  = 4 * DMODEL;
    const int T   = TSEQ;
    const int BH  = BATCH * NHEAD;

    // workspace layout (floats)
    float* ws = (float*)d_ws;
    long off = 0;
    auto alloc = [&](long n) { float* p = ws + off; off += (n + 255) & ~255L; return p; };
    float* vis1  = alloc((long)BATCH * TIMG * 1088);
    float* vis2  = alloc((long)BATCH * TIMG * D);
    float* visln = alloc((long)BATCH * TIMG * D);
    float* h     = alloc((long)M * D);
    float* zbuf  = alloc((long)M * D);
    float* qkv   = alloc((long)M * D3);
    float* probs = alloc((long)BH * T * T);
    float* obuf  = alloc((long)M * D);
    float* abuf  = alloc((long)M * D4);
    float* tmp16 = alloc((long)M * RLORA);

    // ---- visual projection ----
    run_gemm(stream, imgf, vp1_w, vp1_b, nullptr, nullptr, nullptr, vis1,
             BATCH * TIMG, 1088, 1408, 1408, 1088, 1088, 0, 1.f, 1, 1, 0, 0, 0, 0, 0, 0);
    run_gemm(stream, vis1, vp2_w, vp2_b, nullptr, nullptr, nullptr, vis2,
             BATCH * TIMG, D, 1088, 1088, D, D, 0, 1.f, 1, 1, 0, 0, 0, 0, 0, 0);
    ln_kernel<<<BATCH * TIMG, 256, 0, stream>>>(vis2, ln_g, ln_b, visln, D);

    // ---- embeddings ----
    {
        long total = (long)M * D;
        assemble_h<<<(unsigned)((total + 255) / 256), 256, 0, stream>>>(visln, x, wte, wpe, h);
    }

    const float inv_sqrt_hd = 0.125f;  // 1/sqrt(64)

    for (int l = 0; l < LNUM; ++l) {
        const float* l1g = ln1_g + (long)l * D;
        const float* l1b = ln1_b + (long)l * D;
        const float* l2g = ln2_g + (long)l * D;
        const float* l2b = ln2_b + (long)l * D;
        const float* aw  = attn_w + (long)l * D * D3;
        const float* ab  = attn_b + (long)l * D3;
        const float* pw  = ap_w + (long)l * D * D;
        const float* pb  = ap_b + (long)l * D;
        const float* fw  = fc_w + (long)l * D * D4;
        const float* fb  = fc_b + (long)l * D4;
        const float* fA  = fc_A + (long)l * RLORA * D;
        const float* fB  = fc_B + (long)l * D4 * RLORA;
        const float* ow  = pj_w + (long)l * D4 * D;
        const float* ob  = pj_b + (long)l * D;
        const float* oA  = pj_A + (long)l * RLORA * D4;
        const float* oB  = pj_B + (long)l * D * RLORA;
        float* atts_l = atts + (long)l * BH * T * T;

        // z = LN1(h)
        ln_kernel<<<M, 256, 0, stream>>>(h, l1g, l1b, zbuf, D);
        // qkv = z @ attn_w + attn_b
        run_gemm(stream, zbuf, aw, ab, nullptr, nullptr, nullptr, qkv,
                 M, D3, D, D, D3, D3, 0, 1.f, 1, 1, 0, 0, 0, 0, 0, 0);
        // scores = q @ k^T * inv_sqrt  (written to output atts region, pre-mask)
        run_gemm(stream, qkv, qkv + D, nullptr, nullptr, nullptr, nullptr, atts_l,
                 T, T, HDIM, D3, D3, T, 1, inv_sqrt_hd, BH, NHEAD,
                 (long)T * D3, HDIM, (long)T * D3, HDIM,
                 (long)NHEAD * T * T, (long)T * T);
        // probs = softmax(causal(scores))
        {
            dim3 g(T, BH);
            softmax_causal<<<g, 256, 0, stream>>>(atts_l, probs, T);
        }
        // o = probs @ v
        run_gemm(stream, probs, qkv + 2 * D, nullptr, nullptr, nullptr, nullptr, obuf,
                 T, HDIM, T, T, D3, D, 0, 1.f, BH, NHEAD,
                 (long)NHEAD * T * T, (long)T * T, (long)T * D3, HDIM,
                 (long)T * D, HDIM);
        // h = h + o @ ap_w + ap_b
        run_gemm(stream, obuf, pw, pb, h, nullptr, nullptr, h,
                 M, D, D, D, D, D, 0, 1.f, 1, 1, 0, 0, 0, 0, 0, 0);
        // z2 = LN2(h)
        ln_kernel<<<M, 256, 0, stream>>>(h, l2g, l2b, zbuf, D);
        // LoRA down for fc:  tmp16 = z2 @ fc_A^T
        lora_down<<<(M * RLORA + 255) / 256, 256, 0, stream>>>(zbuf, fA, tmp16, M, D);
        // a = z2 @ fc_w + fc_b + lora
        run_gemm(stream, zbuf, fw, fb, nullptr, tmp16, fB, abuf,
                 M, D4, D, D, D4, D4, 0, 1.f, 1, 1, 0, 0, 0, 0, 0, 0);
        // gelu
        {
            long n = (long)M * D4;
            gelu_kernel<<<(unsigned)((n + 255) / 256), 256, 0, stream>>>(abuf, n);
        }
        // LoRA down for proj: tmp16 = a @ pj_A^T
        lora_down<<<(M * RLORA + 255) / 256, 256, 0, stream>>>(abuf, oA, tmp16, M, D4);
        // h = h + a @ pj_w + pj_b + lora
        run_gemm(stream, abuf, ow, ob, h, tmp16, oB, h,
                 M, D, D4, D4, D, D, 0, 1.f, 1, 1, 0, 0, 0, 0, 0, 0);
    }

    // ---- final LN + lm_head (tied wte, transposed) + head LoRA ----
    ln_kernel<<<M, 256, 0, stream>>>(h, lnf_g, lnf_b, zbuf, D);
    lora_down<<<(M * RLORA + 255) / 256, 256, 0, stream>>>(zbuf, head_A, tmp16, M, D);
    run_gemm(stream, zbuf, wte, nullptr, nullptr, tmp16, head_B, logits,
             M, VOCAB, D, D, D, VOCAB, 1, 1.f, 1, 1, 0, 0, 0, 0, 0, 0);
}
